// CharModel_17231408791613
// MI455X (gfx1250) — compile-verified, matrix operand order checked by reading, not run on hardware
//
#include <hip/hip_runtime.h>

// ---------------------------------------------------------------------------
// GRU char model for MI455X (gfx1250, wave32, WMMA).
//   out[b,s,:] = h[b,s] @ W_out + b_out,  h = GRU(embed[x], W_ih, W_hh, ...)
// Persistent-recurrence kernel: 8 WGs x 512 threads (16 waves), one WG per
// 16-row batch tile; all 512 timesteps in-loop with h kept on-chip.
// W_hh streams from L2 each step (too big for VGPRs/LDS); W_ih & W_out
// fragments stay register-resident; h round-trips an 8 KB LDS tile.
// ---------------------------------------------------------------------------

#define VOCAB  256
#define EMBED  16
#define HIDDEN 256
#define BATCH  128
#define SEQ    512
#define GATES  768      // 3*HIDDEN, [r|z|n]

typedef __attribute__((ext_vector_type(16))) __bf16 v16bf;
typedef __attribute__((ext_vector_type(8)))  __bf16 v8bf;
typedef __attribute__((ext_vector_type(8)))  float  v8f;

#define NCHUNKS 16            // 256/16 column chunks (hidden and vocab)
#define KSTEPS  8             // 256/32 k-slices per 256-deep reduction
#define FRAG_ELEMS 512        // one 32x16 bf16 B-fragment: 32 lanes x 16 vals

// workspace layout (bf16 element counts)
#define WHH_FRAG_COUNT (3 * NCHUNKS * KSTEPS * FRAG_ELEMS)  // 196608 (384 KB)
#define WIH_FRAG_COUNT (3 * NCHUNKS * FRAG_ELEMS)           //  24576 ( 48 KB)
#define WOUT_FRAG_COUNT (NCHUNKS * KSTEPS * FRAG_ELEMS)     //  65536 (128 KB)

#define HSTR 264              // padded LDS row stride for h tile (bank spread)

__device__ __forceinline__ int frag_k(int lane, int e) {
  // K position held by (lane, element e) of a 16-bit WMMA operand fragment
  // (A 16x32 / B 32x16 layout, CDNA5 ISA 7.12.2).
  if (lane < 16) return (e < 8) ? e : e + 8;   // K in {0..7, 16..23}
  return (e < 8) ? e + 8 : e + 16;             // K in {8..15, 24..31}
}

__device__ __forceinline__ v8f splat8(float v) {
  v8f r;
#pragma unroll
  for (int i = 0; i < 8; ++i) r[i] = v;
  return r;
}

__device__ __forceinline__ v8f wmma_bf16(v16bf a, v16bf b, v8f c) {
  // D = A(16x32 bf16) x B(32x16 bf16) + C(16x16 f32)
  return __builtin_amdgcn_wmma_f32_16x16x32_bf16(
      /*neg_a=*/false, a, /*neg_b=*/false, b,
      /*c_mod=*/(short)0, c, /*reuse_a=*/false, /*reuse_b=*/false);
}

// ---------------------------------------------------------------------------
// Prep: swizzle f32 weights into bf16 WMMA B-fragment layout in workspace.
// Hot loop then issues pure contiguous 32B per-lane loads for every fragment.
// ---------------------------------------------------------------------------
__global__ void gru_prep_frags(const float* __restrict__ W_ih,
                               const float* __restrict__ W_hh,
                               const float* __restrict__ W_out,
                               __bf16* __restrict__ whh_frag,
                               __bf16* __restrict__ wih_frag,
                               __bf16* __restrict__ wout_frag) {
  int tid = blockIdx.x * blockDim.x + threadIdx.x;
  if (tid < WHH_FRAG_COUNT) {
    // index = ((g*16+m)*8 + k)*512 + lane*16 + e
    int e = tid & 15, lane = (tid >> 4) & 31;
    int k = (tid >> 9) & 7, m = (tid >> 12) & 15, g = tid >> 16;
    int K = frag_k(lane, e);
    int col = g * HIDDEN + m * 16 + (lane & 15);
    int row = k * 32 + K;
    whh_frag[tid] = (__bf16)W_hh[row * GATES + col];
    return;
  }
  tid -= WHH_FRAG_COUNT;
  if (tid < WIH_FRAG_COUNT) {
    // index = (g*16+m)*512 + lane*16 + e ; K padded 16 -> 32 with zeros
    int e = tid & 15, lane = (tid >> 4) & 31;
    int m = (tid >> 9) & 15, g = tid >> 13;
    int K = frag_k(lane, e);
    int col = g * HIDDEN + m * 16 + (lane & 15);
    wih_frag[tid] = (K < EMBED) ? (__bf16)W_ih[K * GATES + col] : (__bf16)0.0f;
    return;
  }
  tid -= WIH_FRAG_COUNT;
  if (tid < WOUT_FRAG_COUNT) {
    // index = (m*8 + k)*512 + lane*16 + e
    int e = tid & 15, lane = (tid >> 4) & 31;
    int k = (tid >> 9) & 7, m = tid >> 12;
    int K = frag_k(lane, e);
    int col = m * 16 + (lane & 15);
    int row = k * 32 + K;
    wout_frag[tid] = (__bf16)W_out[row * VOCAB + col];
  }
}

// ---------------------------------------------------------------------------
// Main persistent GRU kernel. grid = 8 (batch tiles of 16), block = 512.
// Wave w owns hidden/vocab column chunk w for all gates + output projection.
// ---------------------------------------------------------------------------
__global__ void __launch_bounds__(512, 1)
gru_char_model(const int* __restrict__ x,
               const float* __restrict__ embed,
               const float* __restrict__ b_ih,
               const float* __restrict__ b_hh,
               const float* __restrict__ b_out,
               const __bf16* __restrict__ whh_frag,
               const __bf16* __restrict__ wih_frag,
               const __bf16* __restrict__ wout_frag,
               float* __restrict__ out) {
  __shared__ __attribute__((aligned(16))) __bf16 emb_lds[VOCAB * EMBED]; // 8 KB
  __shared__ __attribute__((aligned(16))) __bf16 h_lds[16 * HSTR];       // 8.25 KB

  const int tid  = threadIdx.x;
  const int lane = tid & 31;
  const int wave = tid >> 5;          // 0..15 == column chunk m
  const int lrow = lane & 15;
  const bool hi  = lane >= 16;
  const int b0   = blockIdx.x * 16;   // batch tile origin

  // stage embedding table as bf16 in LDS; zero h(0)
  for (int i = tid; i < VOCAB * EMBED; i += 512) emb_lds[i] = (__bf16)embed[i];
  for (int i = tid; i < 16 * HSTR; i += 512) h_lds[i] = (__bf16)0.0f;

  // per-lane biases for this wave's 16-column chunk (column = C-frag lane)
  const int gcol = wave * 16 + lrow;
  const float bias_r  = b_ih[gcol] + b_hh[gcol];
  const float bias_z  = b_ih[HIDDEN + gcol] + b_hh[HIDDEN + gcol];
  const float bias_ni = b_ih[2 * HIDDEN + gcol];   // kept separate: n-gate
  const float bias_nh = b_hh[2 * HIDDEN + gcol];   // needs tanh(ni + r*nh)
  const float bias_o  = b_out[gcol];

  // register-resident loop-invariant fragments: W_ih (3) and W_out (8)
  const v16bf fihR = *(const v16bf*)&wih_frag[(0 * NCHUNKS + wave) * FRAG_ELEMS + lane * 16];
  const v16bf fihZ = *(const v16bf*)&wih_frag[(1 * NCHUNKS + wave) * FRAG_ELEMS + lane * 16];
  const v16bf fihN = *(const v16bf*)&wih_frag[(2 * NCHUNKS + wave) * FRAG_ELEMS + lane * 16];
  v16bf fout[KSTEPS];
#pragma unroll
  for (int k = 0; k < KSTEPS; ++k)
    fout[k] = *(const v16bf*)&wout_frag[(wave * KSTEPS + k) * FRAG_ELEMS + lane * 16];

  // W_hh streams from L2 every step. Pointer chain stays intact (kernarg
  // provenance => address space 1 => global_load_b128, LOADcnt only); the
  // always-zero 'licm_sink' offset is made opaque to the compiler each
  // iteration so the 24 fragment loads cannot be hoisted (no hoist -> no
  // 192-VGPR spill to scratch, as seen in round 1).
  const __bf16* pR = whh_frag + (0 * NCHUNKS + wave) * (KSTEPS * FRAG_ELEMS) + lane * 16;
  const __bf16* pZ = whh_frag + (1 * NCHUNKS + wave) * (KSTEPS * FRAG_ELEMS) + lane * 16;
  const __bf16* pN = whh_frag + (2 * NCHUNKS + wave) * (KSTEPS * FRAG_ELEMS) + lane * 16;
  int licm_sink = 0;

  __syncthreads();

  v8f hold = splat8(0.0f);       // this wave's h columns, C-fragment layout

  for (int s = 0; s < SEQ; ++s) {
    // ---- input-side A fragment: embedding rows, K padded 16 -> 32 ----
    const int tok = x[(b0 + lrow) * SEQ + s];
    v8bf ec = *(const v8bf*)&emb_lds[tok * EMBED + (hi ? 8 : 0)];
    v16bf ef;
#pragma unroll
    for (int e = 0; e < 8; ++e) { ef[e] = ec[e]; ef[8 + e] = (__bf16)0.0f; }

    // ---- gate pre-activations: bias + emb@W_ih + h@W_hh (WMMA) ----
    v8f ar = splat8(bias_r), az = splat8(bias_z);
    v8f ani = splat8(bias_ni), anh = splat8(bias_nh);
    ar  = wmma_bf16(ef, fihR, ar);
    az  = wmma_bf16(ef, fihZ, az);
    ani = wmma_bf16(ef, fihN, ani);
#pragma unroll
    for (int k = 0; k < KSTEPS; ++k) {
      // h(s) A-fragment k-slice, on demand from LDS (keeps VGPR count low)
      const int abase = lrow * HSTR + k * 32 + (hi ? 8 : 0);
      v8bf a0 = *(const v8bf*)&h_lds[abase];
      v8bf a1 = *(const v8bf*)&h_lds[abase + 16];
      v16bf ah;
#pragma unroll
      for (int e = 0; e < 8; ++e) { ah[e] = a0[e]; ah[8 + e] = a1[e]; }
      const int fo = k * FRAG_ELEMS + licm_sink;
      ar  = wmma_bf16(ah, *(const v16bf*)&pR[fo], ar);
      az  = wmma_bf16(ah, *(const v16bf*)&pZ[fo], az);
      anh = wmma_bf16(ah, *(const v16bf*)&pN[fo], anh);
    }

    // ---- GRU cell, entirely in C-fragment registers ----
    v8f hnew;
#pragma unroll
    for (int i = 0; i < 8; ++i) {
      float r = 1.0f / (1.0f + __expf(-ar[i]));
      float z = 1.0f / (1.0f + __expf(-az[i]));
      float p = ani[i] + r * anh[i];
      float e2 = __expf(-2.0f * p);
      float n = (1.0f - e2) / (1.0f + e2);         // tanh(p)
      hnew[i] = (1.0f - z) * n + z * hold[i];
    }
    hold = hnew;

    __syncthreads();   // all waves done reading h_lds(s)
    {
      // scatter this wave's bf16 h(s+1) slice into LDS (C layout -> row major)
      const int col = wave * 16 + lrow;
#pragma unroll
      for (int r = 0; r < 8; ++r) {
        const int row = r + (hi ? 8 : 0);
        h_lds[row * HSTR + col] = (__bf16)hnew[r];
      }
    }
    __syncthreads();   // h(s+1) visible

    // ---- fused output projection: out[:, s, chunk] = h(s+1)@W_out + b ----
    v8f ao = splat8(bias_o);
#pragma unroll
    for (int k = 0; k < KSTEPS; ++k) {
      const int abase = lrow * HSTR + k * 32 + (hi ? 8 : 0);
      v8bf a0 = *(const v8bf*)&h_lds[abase];
      v8bf a1 = *(const v8bf*)&h_lds[abase + 16];
      v16bf ah;
#pragma unroll
      for (int e = 0; e < 8; ++e) { ah[e] = a0[e]; ah[8 + e] = a1[e]; }
      ao = wmma_bf16(ah, fout[k], ao);
    }
    {
      const int col = wave * 16 + lrow;
#pragma unroll
      for (int r = 0; r < 8; ++r) {
        const int row = r + (hi ? 8 : 0);
        out[(((size_t)(b0 + row)) * SEQ + s) * VOCAB + col] = ao[r];
      }
    }

    // opaque no-op update of the always-zero offset: blocks LICM without
    // breaking global address-space inference on the weight pointers
    asm volatile("" : "+v"(licm_sink));
  }
}

// ---------------------------------------------------------------------------
extern "C" void kernel_launch(void* const* d_in, const int* in_sizes, int n_in,
                              void* d_out, int out_size, void* d_ws, size_t ws_size,
                              hipStream_t stream) {
  (void)in_sizes; (void)n_in; (void)out_size; (void)ws_size;
  const int*   x     = (const int*)  d_in[0];
  const float* embed = (const float*)d_in[1];
  const float* W_ih  = (const float*)d_in[2];
  const float* b_ih  = (const float*)d_in[3];
  const float* W_hh  = (const float*)d_in[4];
  const float* b_hh  = (const float*)d_in[5];
  const float* W_out = (const float*)d_in[6];
  const float* b_out = (const float*)d_in[7];
  float* out = (float*)d_out;

  __bf16* whh_frag  = (__bf16*)d_ws;                 // 384 KB
  __bf16* wih_frag  = whh_frag + WHH_FRAG_COUNT;     //  48 KB
  __bf16* wout_frag = wih_frag + WIH_FRAG_COUNT;     // 128 KB

  const int prep_total = WHH_FRAG_COUNT + WIH_FRAG_COUNT + WOUT_FRAG_COUNT;
  gru_prep_frags<<<(prep_total + 255) / 256, 256, 0, stream>>>(
      W_ih, W_hh, W_out, whh_frag, wih_frag, wout_frag);

  gru_char_model<<<BATCH / 16, 512, 0, stream>>>(
      x, embed, b_ih, b_hh, b_out, whh_frag, wih_frag, wout_frag, out);
}
